// AttentionWithMSR_50130858279883
// MI455X (gfx1250) — matrix operand order repaired and checked by reading, not compiled, not run on hardware
//
#include <hip/hip_runtime.h>
#include <math.h>

typedef __attribute__((ext_vector_type(16))) _Float16 v16h;
typedef __attribute__((ext_vector_type(8)))  _Float16 v8h;
typedef __attribute__((ext_vector_type(8)))  float    v8f;

namespace {
constexpr int    kHW       = 256;
constexpr int    kImgs     = 256;                       // B*C = 4*64
constexpr int    kImgElems = kHW * kHW;                 // 65536
constexpr size_t kN        = (size_t)kImgs * kImgElems; // 16,777,216

// ---- workspace layout (bytes) ----
constexpr size_t SZ_H = kN * 2;          // f16 image-set (33.5 MB)
constexpr size_t SZ_F = kN * 4;          // f32 image-set (67 MB)
constexpr size_t SZ_G = 3 * 65536 * 2;   // f16 G array, 3 scales

constexpr size_t OFF_XAHI  = 0;                     // x, A-fragment layout, f16 hi
constexpr size_t OFF_XALO  = OFF_XAHI + SZ_H;       // f16 lo
constexpr size_t OFF_TMPHI = OFF_XALO + SZ_H;       // pass1 out, B-fragment layout
constexpr size_t OFF_TMPLO = OFF_TMPHI + SZ_H;
constexpr size_t OFF_ACC   = OFF_TMPLO + SZ_H;      // 3*log1p(x) - sum log1p(blur)
constexpr size_t OFF_GAHI  = OFF_ACC + SZ_F;        // G, A-fragment layout
constexpr size_t OFF_GALO  = OFF_GAHI + SZ_G;
constexpr size_t OFF_GBHI  = OFF_GALO + SZ_G;       // G, B-fragment layout
constexpr size_t OFF_GBLO  = OFF_GBHI + SZ_G;
constexpr size_t OFF_NORM  = OFF_GBLO + SZ_G;       // 3 inv-norms (padded 256B)
constexpr size_t OFF_WG    = OFF_NORM + 256;        // folded W_g  (32x64 f32)
constexpr size_t OFF_WX    = OFF_WG + 2048 * 4;     // folded W_x
constexpr size_t OFF_BS    = OFF_WX + 2048 * 4;     // summed folded biases (32)
constexpr size_t OFF_PW    = OFF_BS + 32 * 4;       // folded psi weights (32)
constexpr size_t OFF_PB    = OFF_PW + 32 * 4;       // folded psi bias (1)
}

__device__ __forceinline__ v8f wmma16(v16h a, v16h b, v8f c) {
  return __builtin_amdgcn_wmma_f32_16x16x32_f16(false, a, false, b, (short)0, c,
                                                false, false);
}

// ---------------- fold BatchNorm into the 1x1 conv weights ----------------
__global__ void k_prep_weights(const float* wg_w, const float* wg_b,
                               const float* wg_g, const float* wg_be,
                               const float* wg_m, const float* wg_v,
                               const float* wx_w, const float* wx_b,
                               const float* wx_g, const float* wx_be,
                               const float* wx_m, const float* wx_v,
                               const float* ps_w, const float* ps_b,
                               const float* ps_g, const float* ps_be,
                               const float* ps_m, const float* ps_v,
                               float* Wg, float* Wx, float* biasSum,
                               float* Pw, float* psiBias) {
  int tid = threadIdx.x;
  for (int i = tid; i < 2048; i += 256) {
    int o = i >> 6;
    float sg = wg_g[o] * rsqrtf(wg_v[o] + 1e-5f);
    float sx = wx_g[o] * rsqrtf(wx_v[o] + 1e-5f);
    Wg[i] = wg_w[i] * sg;
    Wx[i] = wx_w[i] * sx;
  }
  if (tid < 32) {
    int o = tid;
    float sg = wg_g[o] * rsqrtf(wg_v[o] + 1e-5f);
    float sx = wx_g[o] * rsqrtf(wx_v[o] + 1e-5f);
    biasSum[o] = sg * (wg_b[o] - wg_m[o]) + wg_be[o] +
                 sx * (wx_b[o] - wx_m[o]) + wx_be[o];
    float sp = ps_g[0] * rsqrtf(ps_v[0] + 1e-5f);
    Pw[o] = sp * ps_w[o];
    if (o == 0) psiBias[0] = sp * (ps_b[0] - ps_m[0]) + ps_be[0];
  }
}

// ---------------- Gaussian normalization (sum over full k taps) ----------------
__global__ void k_norms(float* invNorm) {
  __shared__ float red[256];
  const int SC[3] = {15, 80, 250};
  int scale = SC[blockIdx.x];
  int p = 2 * scale;                       // k = 4*scale+1, p = k/2
  float inv2s2 = 1.0f / (2.0f * (float)scale * (float)scale);
  float acc = 0.0f;
  for (int t = -p + (int)threadIdx.x; t <= p; t += 256)
    acc += expf(-(float)(t * t) * inv2s2);
  red[threadIdx.x] = acc;
  __syncthreads();
  for (int st = 128; st > 0; st >>= 1) {
    if ((int)threadIdx.x < st) red[threadIdx.x] += red[threadIdx.x + st];
    __syncthreads();
  }
  if (threadIdx.x == 0) invNorm[blockIdx.x] = 1.0f / red[0];
}

// ---------------- build G (256x256 banded Gaussian matrix) in fragment layouts ----
// A-frag (16x32, MxK): lane l, elem e -> m=l&15, k = e + 16*(l>>4)
// B-frag (32x16, KxN): lane l, elem e -> n=l&15, k = e + 16*(l>>4)
__global__ void k_build_G(const float* invNorm, _Float16* GAhi, _Float16* GAlo,
                          _Float16* GBhi, _Float16* GBlo) {
  const int SC[3] = {15, 80, 250};
  int w    = blockIdx.x * 8 + (threadIdx.x >> 5);   // 0..767
  int lane = threadIdx.x & 31;
  bool isA = w < 384;
  int wi = isA ? w : (w - 384);
  int s  = wi >> 7;                                  // scale index
  int tl = wi & 127;                                 // tile index (128 per layout)
  int scale = SC[s];
  int p = 2 * scale;
  float inv2s2 = 1.0f / (2.0f * (float)scale * (float)scale);
  float inorm  = invNorm[s];
  size_t base = (size_t)s * 65536 + (size_t)tl * 512 + (size_t)lane * 16;
  #pragma unroll
  for (int e = 0; e < 16; e++) {
    int row, col;
    if (isA) {                       // tl = Mt*8 + Kt ; row=i_out, col=i'
      int m = lane & 15;
      int k = e + ((lane >> 4) << 4);
      row = (tl >> 3) * 16 + m;
      col = (tl & 7) * 32 + k;
    } else {                         // tl = KBt*16 + NBt ; row=j', col=j
      int n = lane & 15;
      int k = e + ((lane >> 4) << 4);
      row = (tl >> 4) * 32 + k;
      col = (tl & 15) * 16 + n;
    }
    int d = col - row;
    float v = 0.0f;
    if (d >= -p && d <= p) v = expf(-(float)(d * d) * inv2s2) * inorm;
    _Float16 hi = (_Float16)v;
    _Float16 lo = (_Float16)(v - (float)hi);
    if (isA) { GAhi[base + e] = hi; GAlo[base + e] = lo; }
    else     { GBhi[base + e] = hi; GBlo[base + e] = lo; }
  }
}

// ---------------- swizzle x into A-fragment layout (hi/lo) + seed acc --------
__global__ void k_prep_x(const float* x, _Float16* XAhi, _Float16* XAlo,
                         float* acc) {
  int w    = blockIdx.x * 8 + (threadIdx.x >> 5);   // 0..32767
  int lane = threadIdx.x & 31;
  int img = w >> 7;
  int tl  = w & 127;                                 // Mt*8 + Kt
  int Mt = tl >> 3, Kt = tl & 7;
  int m = lane & 15;
  size_t ibase = (size_t)img * 65536;
  size_t obase = ibase + (size_t)tl * 512 + (size_t)lane * 16;
  #pragma unroll
  for (int e = 0; e < 16; e++) {
    int k = e + ((lane >> 4) << 4);
    int i = Mt * 16 + m, j = Kt * 32 + k;
    size_t idx = ibase + (size_t)i * 256 + j;
    float v = x[idx];
    _Float16 hi = (_Float16)v;
    XAhi[obase + e] = hi;
    XAlo[obase + e] = (_Float16)(v - (float)hi);
    acc[idx] = 3.0f * log1pf(v);
  }
}

// ---------------- pass 1: TMP = X * G (horizontal blur) ----------------------
// wave: 1 M-tile x 4 N-tiles; output written in B-fragment layout for pass 2.
__global__ void k_pass1(const _Float16* __restrict__ XAhi,
                        const _Float16* __restrict__ XAlo,
                        const _Float16* __restrict__ GBhi,
                        const _Float16* __restrict__ GBlo,
                        _Float16* __restrict__ Thi, _Float16* __restrict__ Tlo,
                        int s) {
  int w    = blockIdx.x * 8 + (threadIdx.x >> 5);   // 0..16383
  int lane = threadIdx.x & 31;
  int img = w >> 6;
  int rem = w & 63;
  int Mt  = rem >> 2;
  int Nt0 = (rem & 3) * 4;
  size_t ib = (size_t)img * 65536;
  size_t gb = (size_t)s * 65536;
  size_t laneOff = (size_t)lane * 16;

  v8f acc[4];
  #pragma unroll
  for (int t = 0; t < 4; t++)
    #pragma unroll
    for (int r = 0; r < 8; r++) acc[t][r] = 0.0f;

  #pragma unroll 1
  for (int kt = 0; kt < 8; kt++) {
    size_t aoff = ib + (size_t)(Mt * 8 + kt) * 512 + laneOff;
    v16h ahi = *(const v16h*)(XAhi + aoff);
    v16h alo = *(const v16h*)(XAlo + aoff);
    #pragma unroll
    for (int t = 0; t < 4; t++) {
      size_t boff = gb + (size_t)(kt * 16 + Nt0 + t) * 512 + laneOff;
      v16h bhi = *(const v16h*)(GBhi + boff);
      v16h blo = *(const v16h*)(GBlo + boff);
      acc[t] = wmma16(ahi, bhi, acc[t]);   // hi*hi
      acc[t] = wmma16(ahi, blo, acc[t]);   // hi*lo
      acc[t] = wmma16(alo, bhi, acc[t]);   // lo*hi
    }
  }

  // scatter f32 accumulators into pass-2 B-fragment layout (contiguous v8h)
  int KBt   = Mt >> 1;
  int lanep = (lane & 15) + ((Mt & 1) << 4);
  int esub  = (lane >> 4) * 8;
  #pragma unroll
  for (int t = 0; t < 4; t++) {
    size_t off = ib + (size_t)(KBt * 16 + Nt0 + t) * 512 +
                 (size_t)lanep * 16 + esub;
    v8h hi, lo;
    #pragma unroll
    for (int r = 0; r < 8; r++) {
      float v = acc[t][r];
      _Float16 h = (_Float16)v;
      hi[r] = h;
      lo[r] = (_Float16)(v - (float)h);
    }
    *(v8h*)(Thi + off) = hi;
    *(v8h*)(Tlo + off) = lo;
  }
}

// ---------------- pass 2: BLUR = G * TMP ; acc -= log1p(BLUR) ----------------
__global__ void k_pass2(const _Float16* __restrict__ GAhi,
                        const _Float16* __restrict__ GAlo,
                        const _Float16* __restrict__ Thi,
                        const _Float16* __restrict__ Tlo,
                        float* __restrict__ acc, int s) {
  int w    = blockIdx.x * 8 + (threadIdx.x >> 5);
  int lane = threadIdx.x & 31;
  int img = w >> 6;
  int rem = w & 63;
  int Mt  = rem >> 2;
  int Nt0 = (rem & 3) * 4;
  size_t ib = (size_t)img * 65536;
  size_t gb = (size_t)s * 65536;
  size_t laneOff = (size_t)lane * 16;

  v8f d[4];
  #pragma unroll
  for (int t = 0; t < 4; t++)
    #pragma unroll
    for (int r = 0; r < 8; r++) d[t][r] = 0.0f;

  #pragma unroll 1
  for (int kt = 0; kt < 8; kt++) {
    size_t aoff = gb + (size_t)(Mt * 8 + kt) * 512 + laneOff;
    v16h ahi = *(const v16h*)(GAhi + aoff);
    v16h alo = *(const v16h*)(GAlo + aoff);
    #pragma unroll
    for (int t = 0; t < 4; t++) {
      size_t boff = ib + (size_t)(kt * 16 + Nt0 + t) * 512 + laneOff;
      v16h bhi = *(const v16h*)(Thi + boff);
      v16h blo = *(const v16h*)(Tlo + boff);
      d[t] = wmma16(ahi, bhi, d[t]);
      d[t] = wmma16(ahi, blo, d[t]);
      d[t] = wmma16(alo, bhi, d[t]);
    }
  }

  int n  = lane & 15;
  int mh = (lane >> 4) * 8;
  #pragma unroll
  for (int t = 0; t < 4; t++) {
    int j = (Nt0 + t) * 16 + n;
    #pragma unroll
    for (int r = 0; r < 8; r++) {
      int i = Mt * 16 + mh + r;
      size_t idx = ib + (size_t)i * 256 + j;
      acc[idx] -= log1pf(d[t][r]);
    }
  }
}

// ---------------- fused attention gate: convs + BN + relu + psi + gating -----
__global__ void k_final(const float* __restrict__ g, const float* __restrict__ x,
                        const float* __restrict__ accArr,
                        const float* __restrict__ Wg, const float* __restrict__ Wx,
                        const float* __restrict__ biasSum,
                        const float* __restrict__ Pw,
                        const float* __restrict__ psiBias,
                        float* __restrict__ out) {
  __shared__ float sWg[2048];
  __shared__ float sWx[2048];
  __shared__ float sB[32];
  __shared__ float sP[32];
  __shared__ float sPB;
  int tid = threadIdx.x;
  for (int i = tid; i < 2048; i += 256) { sWg[i] = Wg[i]; sWx[i] = Wx[i]; }
  if (tid < 32) { sB[tid] = biasSum[tid]; sP[tid] = Pw[tid]; }
  if (tid == 0) sPB = psiBias[0];
  __syncthreads();

  int pix = blockIdx.x * 256 + tid;      // 0 .. 262143
  int b   = pix >> 16;
  int hw  = pix & 65535;
  size_t base = (size_t)b * 64 * 65536 + hw;

  float a[32];
  #pragma unroll
  for (int o = 0; o < 32; o++) a[o] = sB[o];

  #pragma unroll 1
  for (int c = 0; c < 64; c++) {
    float gv = g[base + (size_t)c * 65536];
    float mv = accArr[base + (size_t)c * 65536] * (1.0f / 3.0f);
    #pragma unroll
    for (int o = 0; o < 32; o++)
      a[o] = fmaf(sWg[o * 64 + c], gv, fmaf(sWx[o * 64 + c], mv, a[o]));
  }

  float z = sPB;
  #pragma unroll
  for (int o = 0; o < 32; o++) z = fmaf(sP[o], fmaxf(a[o], 0.0f), z);
  float psi = 1.0f / (1.0f + expf(-z));

  #pragma unroll 1
  for (int c = 0; c < 64; c++) {
    size_t idx = base + (size_t)c * 65536;
    out[idx] = x[idx] * psi;
  }
}

extern "C" void kernel_launch(void* const* d_in, const int* in_sizes, int n_in,
                              void* d_out, int out_size, void* d_ws,
                              size_t ws_size, hipStream_t stream) {
  (void)in_sizes; (void)n_in; (void)out_size; (void)ws_size;
  const float* g    = (const float*)d_in[0];
  const float* x    = (const float*)d_in[1];
  const float* wg_w = (const float*)d_in[2];
  const float* wg_b = (const float*)d_in[3];
  const float* wg_g = (const float*)d_in[4];
  const float* wg_be= (const float*)d_in[5];
  const float* wg_m = (const float*)d_in[6];
  const float* wg_v = (const float*)d_in[7];
  const float* wx_w = (const float*)d_in[8];
  const float* wx_b = (const float*)d_in[9];
  const float* wx_g = (const float*)d_in[10];
  const float* wx_be= (const float*)d_in[11];
  const float* wx_m = (const float*)d_in[12];
  const float* wx_v = (const float*)d_in[13];
  const float* ps_w = (const float*)d_in[14];
  const float* ps_b = (const float*)d_in[15];
  const float* ps_g = (const float*)d_in[16];
  const float* ps_be= (const float*)d_in[17];
  const float* ps_m = (const float*)d_in[18];
  const float* ps_v = (const float*)d_in[19];
  float* out = (float*)d_out;

  char* ws = (char*)d_ws;
  _Float16* XAhi = (_Float16*)(ws + OFF_XAHI);
  _Float16* XAlo = (_Float16*)(ws + OFF_XALO);
  _Float16* Thi  = (_Float16*)(ws + OFF_TMPHI);
  _Float16* Tlo  = (_Float16*)(ws + OFF_TMPLO);
  float*    acc  = (float*)   (ws + OFF_ACC);
  _Float16* GAhi = (_Float16*)(ws + OFF_GAHI);
  _Float16* GAlo = (_Float16*)(ws + OFF_GALO);
  _Float16* GBhi = (_Float16*)(ws + OFF_GBHI);
  _Float16* GBlo = (_Float16*)(ws + OFF_GBLO);
  float*    nrm  = (float*)   (ws + OFF_NORM);
  float*    Wg   = (float*)   (ws + OFF_WG);
  float*    Wx   = (float*)   (ws + OFF_WX);
  float*    BS   = (float*)   (ws + OFF_BS);
  float*    Pw   = (float*)   (ws + OFF_PW);
  float*    PB   = (float*)   (ws + OFF_PB);

  k_prep_weights<<<1, 256, 0, stream>>>(wg_w, wg_b, wg_g, wg_be, wg_m, wg_v,
                                        wx_w, wx_b, wx_g, wx_be, wx_m, wx_v,
                                        ps_w, ps_b, ps_g, ps_be, ps_m, ps_v,
                                        Wg, Wx, BS, Pw, PB);
  k_norms<<<3, 256, 0, stream>>>(nrm);
  k_build_G<<<96, 256, 0, stream>>>(nrm, GAhi, GAlo, GBhi, GBlo);
  k_prep_x<<<4096, 256, 0, stream>>>(x, XAhi, XAlo, acc);
  for (int s = 0; s < 3; s++) {
    k_pass1<<<2048, 256, 0, stream>>>(XAhi, XAlo, GBhi, GBlo, Thi, Tlo, s);
    k_pass2<<<2048, 256, 0, stream>>>(GAhi, GAlo, Thi, Tlo, acc, s);
  }
  k_final<<<1024, 256, 0, stream>>>(g, x, acc, Wg, Wx, BS, Pw, PB, out);
}
